// RegionProposalNetwork_30709016167065
// MI455X (gfx1250) — matrix operand order repaired and checked by reading, not compiled, not run on hardware
//
#include <hip/hip_runtime.h>
#include <hip/hip_bf16.h>
#include <stdint.h>
#include <stddef.h>

// ---------------------------------------------------------------------------
// Types for WMMA (gfx1250, wave32)
// ---------------------------------------------------------------------------
typedef __attribute__((ext_vector_type(16))) __bf16        v16bf;
typedef __attribute__((ext_vector_type(8)))  float         v8f;
typedef __attribute__((ext_vector_type(4)))  unsigned int  u32x4;

union FragBF { u32x4 q[2]; v16bf bf; };

#define NEG_INF (-__builtin_inff())

__device__ __forceinline__ unsigned short f2bf(float f) {
  union { float f; unsigned u; } x; x.f = f;
  unsigned r = x.u + 0x7fffu + ((x.u >> 16) & 1u);   // round-to-nearest-even
  return (unsigned short)(r >> 16);
}

// GLOBAL_LOAD_ASYNC_TO_LDS_B128: 16B global -> LDS, tracked by ASYNCcnt.
__device__ __forceinline__ void async_copy_b128(unsigned lds_off, const void* gptr) {
  asm volatile("global_load_async_to_lds_b128 %0, %1, off"
               :: "v"(lds_off), "v"(gptr) : "memory");
}
__device__ __forceinline__ void wait_asynccnt0() {
  asm volatile("s_wait_asynccnt 0x0" ::: "memory");
}
__device__ __forceinline__ unsigned lds_off_u32(const void* p) {
  return (unsigned)(uintptr_t)p;   // shared-aperture flat addr: low 32 bits = LDS offset
}

__device__ __forceinline__ v8f wmma_bf16(const FragBF& a, const FragBF& b, v8f c) {
  return __builtin_amdgcn_wmma_f32_16x16x32_bf16(false, a.bf, false, b.bf,
                                                 (short)0, c, false, false);
}

// ---------------------------------------------------------------------------
// Layout conversions (one-shot, bandwidth-trivial)
// ---------------------------------------------------------------------------
// x: NCHW f32 -> NHWC bf16   dst[(b*2500+p)*512 + c]
__global__ void k_x_to_nhwc_bf16(const float* __restrict__ src,
                                 unsigned short* __restrict__ dst) {
  int g = blockIdx.x * blockDim.x + threadIdx.x;
  if (g >= 4 * 2500 * 512) return;
  int c  = g & 511;
  int bp = g >> 9;               // b*2500 + p
  int b  = bp / 2500, p = bp % 2500;
  dst[g] = f2bf(src[((size_t)(b * 512 + c)) * 2500 + p]);
}

// w: OIHW f32 -> bf16 with k-order k = rs*512 + c  (rs = ry*3+rx)
__global__ void k_w_repack_bf16(const float* __restrict__ src,
                                unsigned short* __restrict__ dst) {
  int g = blockIdx.x * blockDim.x + threadIdx.x;
  if (g >= 512 * 4608) return;
  int n = g / 4608, k = g % 4608;
  int rs = k >> 9, c = k & 511;
  dst[g] = f2bf(src[(size_t)n * 4608 + c * 9 + rs]);
}

// Fused head weight [64 x 512] (18 score ch + 36 loc ch + 10 zero) + bias
// + zero-pad row used by the conv's padding fast path.
__global__ void k_build_head_w(const float* __restrict__ sw, const float* __restrict__ sb,
                               const float* __restrict__ lw, const float* __restrict__ lb,
                               unsigned short* __restrict__ wsl, float* __restrict__ bsl,
                               unsigned short* __restrict__ zrow) {
  int i = blockIdx.x * blockDim.x + threadIdx.x;
  if (i < 64 * 512) {
    int n = i >> 9, c = i & 511;
    float v = 0.f;
    if (n < 18)      v = sw[n * 512 + c];
    else if (n < 54) v = lw[(n - 18) * 512 + c];
    wsl[n * 512 + c] = f2bf(v);
  }
  if (i < 64) {
    float v = 0.f;
    if (i < 18)      v = sb[i];
    else if (i < 54) v = lb[i - 18];
    bsl[i] = v;
  }
  if (i < 128) zrow[i] = 0;      // 256B zero pad row
}

// ---------------------------------------------------------------------------
// 3x3 conv 512->512 as implicit GEMM:  M=10000 pixels, N=512 oc, K=4608
// Block: 256 thr = 8 waves tiled 2(M) x 4(N); block tile 64x128; wave tile
// 32x32 => 4 WMMAs per wave per K-step. Tap-major K loop: the im2col guard,
// tap decode and pixel base address hoist out of the hot loop; padding lanes
// read a zeroed row with stride 0 => unconditional b128 A loads. B tile via
// async-to-LDS. Fused bias+ReLU, writes NHWC bf16 feat.
// ---------------------------------------------------------------------------
__global__ __launch_bounds__(256)
void k_conv3x3_wmma(const unsigned short* __restrict__ xbf,   // NHWC
                    const unsigned short* __restrict__ wbf,   // [oc][rs*512+c]
                    const float* __restrict__ bias,
                    const unsigned short* __restrict__ zrow,  // zeroed pad row
                    unsigned short* __restrict__ featbf) {    // [m][oc]
  __shared__ __align__(16) unsigned short As[64 * 32];
  __shared__ __align__(16) unsigned short Bs[128 * 32];

  const int tid  = threadIdx.x;
  const int lane = tid & 31;
  const int wv   = tid >> 5;
  const int wm   = wv >> 2;      // wave row (M): 0..1
  const int wn   = wv & 3;       // wave col (N): 0..3
  const int m0   = blockIdx.x * 64;
  const int n0   = blockIdx.y * 128;

  v8f acc00 = {}, acc01 = {}, acc10 = {}, acc11 = {};

  // A-tile loader: one pixel row, 8 consecutive c per thread (contiguous NHWC)
  const int miA  = tid & 63;
  const int kbA  = (tid >> 6) * 8;
  const int m    = m0 + miA;
  const int mval = (m < 10000);
  const int bA   = m / 2500;
  const int pA   = m % 2500;
  const int yA   = pA / 50, xA = pA % 50;
  const int rowBase = bA * 2500;
  unsigned short* asDst = &As[miA * 32 + kbA];

  // B-tile loader: one oc row, 16 consecutive k per thread (2 async b128)
  const int njB = tid >> 1;
  const int kbB = (tid & 1) * 16;
  const unsigned short* wrowB = wbf + (size_t)(n0 + njB) * 4608 + kbB;
  const unsigned ldsB0 = lds_off_u32(&Bs[njB * 32 + kbB]);
  const unsigned ldsB1 = ldsB0 + 16;

  // loop-invariant fragment pointers (contiguous k order: lane half h holds k = 16h..16h+15 for A and B)
  const u32x4* paf0 = (const u32x4*)&As[(wm * 32 + (lane & 15)) * 32];
  const u32x4* paf1 = (const u32x4*)&As[(wm * 32 + 16 + (lane & 15)) * 32];
  const u32x4* pbf0 = (const u32x4*)&Bs[(wn * 32 + (lane & 15)) * 32];
  const u32x4* pbf1 = (const u32x4*)&Bs[(wn * 32 + 16 + (lane & 15)) * 32];
  const int qa = (lane < 16) ? 0 : 2;   // A: chunks {0,1} / {2,3}
  const int hb = (lane < 16) ? 0 : 2;   // B: chunks {0,1} / {2,3}

  for (int rs = 0; rs < 9; ++rs) {      // spatial tap (ry,rx)
    const int r  = (rs * 86) >> 8;      // rs / 3
    const int s  = rs - 3 * r;          // rs % 3
    const int yy = yA + r - 1, xx = xA + s - 1;
    const bool av = mval && (unsigned)yy < 50u && (unsigned)xx < 50u;
    const unsigned short* ap =
        av ? (xbf + (((size_t)(rowBase + yy * 50 + xx)) << 9) + kbA) : zrow;
    const int astep = av ? 32 : 0;      // padding lanes re-read the zero row
    const unsigned short* bp = wrowB + rs * 512;

    for (int c0 = 0; c0 < 512; c0 += 32) {
      // B tile: 2x async global->LDS (ASYNCcnt)
      async_copy_b128(ldsB0, bp);
      async_copy_b128(ldsB1, bp + 8);
      __builtin_prefetch(bp + 32, 0, 1);        // global_prefetch_b8

      // A tile: unconditional aligned b128 load
      u32x4 va = *(const u32x4*)ap;
      *(u32x4*)asDst = va;
      ap += astep;
      bp += 32;

      wait_asynccnt0();
      __syncthreads();

      FragBF fa0, fa1, fb0, fb1;
      fa0.q[0] = paf0[qa]; fa0.q[1] = paf0[qa + 1];
      fa1.q[0] = paf1[qa]; fa1.q[1] = paf1[qa + 1];
      fb0.q[0] = pbf0[hb]; fb0.q[1] = pbf0[hb + 1];
      fb1.q[0] = pbf1[hb]; fb1.q[1] = pbf1[hb + 1];

      acc00 = wmma_bf16(fa0, fb0, acc00);
      acc01 = wmma_bf16(fa0, fb1, acc01);
      acc10 = wmma_bf16(fa1, fb0, acc10);
      acc11 = wmma_bf16(fa1, fb1, acc11);
      __syncthreads();
    }
  }

  // ---- epilogue: bias + ReLU, NHWC bf16 store ----
#pragma unroll
  for (int r = 0; r < 8; ++r) {
    int mr0 = m0 + wm * 32 + r + ((lane >= 16) ? 8 : 0);
    int mr1 = mr0 + 16;
    int oc  = n0 + wn * 32 + (lane & 15);
    if (mr0 < 10000) {
      featbf[(size_t)mr0 * 512 + oc]      = f2bf(fmaxf(acc00[r] + bias[oc], 0.f));
      featbf[(size_t)mr0 * 512 + oc + 16] = f2bf(fmaxf(acc01[r] + bias[oc + 16], 0.f));
    }
    if (mr1 < 10000) {
      featbf[(size_t)mr1 * 512 + oc]      = f2bf(fmaxf(acc10[r] + bias[oc], 0.f));
      featbf[(size_t)mr1 * 512 + oc + 16] = f2bf(fmaxf(acc11[r] + bias[oc + 16], 0.f));
    }
  }
}

// ---------------------------------------------------------------------------
// Fused 1x1 head conv (score 18 + loc 36, padded to 64): M=10000, N=64, K=512
// Both A and B tiles are unconditional copies => fully async-to-LDS.
// ---------------------------------------------------------------------------
__global__ __launch_bounds__(256)
void k_head_wmma(const unsigned short* __restrict__ featbf,   // [m][c]
                 const unsigned short* __restrict__ wsl,      // [n][c]
                 const float* __restrict__ bsl,
                 float* __restrict__ head) {                  // [m][64]
  __shared__ __align__(16) unsigned short As[64 * 32];
  __shared__ __align__(16) unsigned short Bs[64 * 32];

  const int tid  = threadIdx.x;
  const int lane = tid & 31;
  const int wv   = tid >> 5;
  const int wm   = wv & 3;
  const int wn   = wv >> 2;
  const int m0   = blockIdx.x * 64;

  v8f acc0 = {}; v8f acc1 = {};

  const int miA    = tid & 63;
  const int kbA    = (tid >> 6) * 8;
  const int m      = m0 + miA;
  const int mclamp = (m < 10000) ? m : 9999;  // rows >=10000 feed only guarded outputs
  const unsigned short* arow = featbf + (size_t)mclamp * 512 + kbA;
  const unsigned ldsA = lds_off_u32(&As[miA * 32 + kbA]);

  const int njB = tid >> 2;
  const int kbB = (tid & 3) * 8;
  const unsigned short* wrow = wsl + (size_t)njB * 512 + kbB;
  const unsigned ldsB = lds_off_u32(&Bs[njB * 32 + kbB]);

  for (int k0 = 0; k0 < 512; k0 += 32) {
    async_copy_b128(ldsA, arow + k0);
    async_copy_b128(ldsB, wrow + k0);
    __builtin_prefetch(arow + k0 + 32, 0, 1);

    wait_asynccnt0();
    __syncthreads();

    FragBF fa;
    const u32x4* pa = (const u32x4*)&As[(wm * 16 + (lane & 15)) * 32];
    fa.q[0] = pa[(lane < 16) ? 0 : 2];
    fa.q[1] = pa[(lane < 16) ? 1 : 3];

    const int hb = (lane < 16) ? 0 : 2;
    FragBF fb0, fb1;
    const u32x4* pb0 = (const u32x4*)&Bs[(wn * 32 + (lane & 15)) * 32];
    const u32x4* pb1 = (const u32x4*)&Bs[(wn * 32 + 16 + (lane & 15)) * 32];
    fb0.q[0] = pb0[hb]; fb0.q[1] = pb0[hb + 1];
    fb1.q[0] = pb1[hb]; fb1.q[1] = pb1[hb + 1];

    acc0 = wmma_bf16(fa, fb0, acc0);
    acc1 = wmma_bf16(fa, fb1, acc1);
    __syncthreads();
  }

#pragma unroll
  for (int r = 0; r < 8; ++r) {
    int mrow = m0 + wm * 16 + r + ((lane >= 16) ? 8 : 0);
    if (mrow < 10000) {
      int n = wn * 32 + (lane & 15);
      head[(size_t)mrow * 64 + n]      = acc0[r] + bsl[n];
      head[(size_t)mrow * 64 + n + 16] = acc1[r] + bsl[n + 16];
    }
  }
}

// ---------------------------------------------------------------------------
// Per-(b,pixel,anchor) postprocess: outputs locs/scores/anchor, and builds
// clipped ROI boxes + min-size-filtered fg score (softmax) for the NMS stage.
// ---------------------------------------------------------------------------
__global__ void k_postproc(const float* __restrict__ head,
                           float* __restrict__ out_locs,    // [4,22500,4]
                           float* __restrict__ out_scores,  // [4,22500,2]
                           float* __restrict__ out_anchor,  // [22500,4]
                           float* __restrict__ roiws,       // [4,22500,4]
                           float* __restrict__ fgsc,        // [4,22500]
                           const int* __restrict__ img_h_p,
                           const int* __restrict__ img_w_p) {
  int g = blockIdx.x * blockDim.x + threadIdx.x;
  if (g >= 4 * 22500) return;
  int b = g / 22500, q = g % 22500, p = q / 9, a = q % 9;

  const float* hp = head + ((size_t)(b * 2500 + p)) * 64;
  float s0 = hp[a * 2 + 0], s1 = hp[a * 2 + 1];
  out_scores[(size_t)(b * 22500 + q) * 2 + 0] = s0;
  out_scores[(size_t)(b * 22500 + q) * 2 + 1] = s1;
  float mx = fmaxf(s0, s1);
  float e0 = __expf(s0 - mx), e1 = __expf(s1 - mx);
  float fg = e1 / (e0 + e1);

  float dy = hp[18 + a * 4 + 0], dx = hp[18 + a * 4 + 1];
  float dh = hp[18 + a * 4 + 2], dw = hp[18 + a * 4 + 3];
  out_locs[(size_t)(b * 22500 + q) * 4 + 0] = dy;
  out_locs[(size_t)(b * 22500 + q) * 4 + 1] = dx;
  out_locs[(size_t)(b * 22500 + q) * 4 + 2] = dh;
  out_locs[(size_t)(b * 22500 + q) * 4 + 3] = dw;

  // anchor (replicating the reference's meshgrid('ij') index quirk)
  const float ratios[3] = {0.5f, 1.0f, 2.0f};
  const float scales[3] = {8.0f, 16.0f, 32.0f};
  int ri = a / 3, si = a % 3;
  float ah = 16.f * scales[si] * __fsqrt_rn(ratios[ri]);
  float aw = 16.f * scales[si] * __fsqrt_rn(1.f / ratios[ri]);
  float ay = 16.f * (float)(p % 50);   // gy = sy[j], j = p % 50
  float ax = 16.f * (float)(p / 50);   // gx = sx[i], i = p / 50
  float y1 = 8.f - ah * 0.5f + ay, x1 = 8.f - aw * 0.5f + ax;
  float y2 = 8.f + ah * 0.5f + ay, x2 = 8.f + aw * 0.5f + ax;
  if (b == 0) {
    out_anchor[(size_t)q * 4 + 0] = y1;
    out_anchor[(size_t)q * 4 + 1] = x1;
    out_anchor[(size_t)q * 4 + 2] = y2;
    out_anchor[(size_t)q * 4 + 3] = x2;
  }

  // loc2bbox + clip + min-size filter
  float H = (float)img_h_p[0], W = (float)img_w_p[0];
  float hh = y2 - y1, ww = x2 - x1;
  float cy = y1 + 0.5f * hh + dy * hh;
  float cx = x1 + 0.5f * ww + dx * ww;
  hh *= __expf(dh);  ww *= __expf(dw);
  float r0 = fminf(fmaxf(cy - 0.5f * hh, 0.f), H);
  float r1 = fminf(fmaxf(cx - 0.5f * ww, 0.f), W);
  float r2 = fminf(fmaxf(cy + 0.5f * hh, 0.f), H);
  float r3 = fminf(fmaxf(cx + 0.5f * ww, 0.f), W);
  size_t ro = (size_t)(b * 22500 + q) * 4;
  roiws[ro + 0] = r0; roiws[ro + 1] = r1; roiws[ro + 2] = r2; roiws[ro + 3] = r3;
  bool valid = (r2 - r0 >= 16.f) && (r3 - r1 >= 16.f);
  fgsc[b * 22500 + q] = valid ? fg : NEG_INF;
}

// ---------------------------------------------------------------------------
// top-k via bitonic sort (descending) over 32768-padded (score, idx) pairs
// ---------------------------------------------------------------------------
__global__ void k_sort_init(const float* __restrict__ fgsc,
                            float* __restrict__ key, int* __restrict__ idx) {
  int g = blockIdx.x * blockDim.x + threadIdx.x;     // 4 * 32768
  if (g >= 4 * 32768) return;
  int b = g >> 15, t = g & 32767;
  key[g] = (t < 22500) ? fgsc[b * 22500 + t] : NEG_INF;
  idx[g] = (t < 22500) ? t : 0;
}

__global__ void k_bitonic(float* __restrict__ key, int* __restrict__ idx,
                          int j, int k) {
  int b = blockIdx.y;
  unsigned i = blockIdx.x * blockDim.x + threadIdx.x;  // 0..32767
  unsigned ixj = i ^ (unsigned)j;
  if (ixj > i) {
    float* kb = key + b * 32768; int* ib = idx + b * 32768;
    float a0 = kb[i], a1 = kb[ixj];
    bool desc = ((i & (unsigned)k) == 0);
    bool sw = desc ? (a0 < a1) : (a0 > a1);
    if (sw) {
      kb[i] = a1; kb[ixj] = a0;
      int t = ib[i]; ib[i] = ib[ixj]; ib[ixj] = t;
    }
  }
}

// ---------------------------------------------------------------------------
// NMS: one workgroup per image; 6000 boxes resident in 96KB dynamic LDS
// (CDNA5: 320KB/WGP). Serial suppression scan with parallel inner loop.
// ---------------------------------------------------------------------------
__global__ __launch_bounds__(1024)
void k_nms(const float* __restrict__ key, const int* __restrict__ idx,
           const float* __restrict__ roiws, float* __restrict__ rois_out) {
  extern __shared__ unsigned char smem[];
  float4* bx = (float4*)smem;                                     // 6000 * 16B
  unsigned char* keep = (unsigned char*)(smem + 6000 * sizeof(float4));
  const int b = blockIdx.x;
  const int tid = threadIdx.x;

  const float4* rp = (const float4*)roiws;
  for (int t = tid; t < 6000; t += blockDim.x) {
    float sc = key[b * 32768 + t];
    int gi = idx[b * 32768 + t];
    bx[t] = rp[b * 22500 + gi];
    keep[t] = (sc > NEG_INF) ? 1 : 0;
  }
  __syncthreads();

  for (int i = 0; i < 6000; ++i) {
    if (keep[i]) {
      float4 bi = bx[i];
      float ai = (bi.z - bi.x) * (bi.w - bi.y);
      for (int jj = i + 1 + tid; jj < 6000; jj += blockDim.x) {
        if (keep[jj]) {
          float4 bj = bx[jj];
          float aj = (bj.z - bj.x) * (bj.w - bj.y);
          float ty = fmaxf(bi.x, bj.x), tx = fmaxf(bi.y, bj.y);
          float by = fminf(bi.z, bj.z), bxr = fminf(bi.w, bj.w);
          float inter = fmaxf(by - ty, 0.f) * fmaxf(bxr - tx, 0.f);
          float iou = inter / (ai + aj - inter);
          if (iou > 0.7f) keep[jj] = 0;
        }
      }
    }
    __syncthreads();
  }

  if (tid == 0) {
    float4* ro = (float4*)rois_out;
    int cnt = 0;
    for (int i = 0; i < 6000 && cnt < 300; ++i)
      if (keep[i]) { ro[b * 300 + cnt] = bx[i]; ++cnt; }
    for (; cnt < 300; ++cnt) ro[b * 300 + cnt] = make_float4(0.f, 0.f, 0.f, 0.f);
  }
}

__global__ void k_roi_indices(int* __restrict__ out) {
  int g = blockIdx.x * blockDim.x + threadIdx.x;
  if (g < 4 * 300) out[g] = g / 300;
}

// ---------------------------------------------------------------------------
// Launch
// ---------------------------------------------------------------------------
extern "C" void kernel_launch(void* const* d_in, const int* in_sizes, int n_in,
                              void* d_out, int out_size, void* d_ws, size_t ws_size,
                              hipStream_t stream) {
  (void)in_sizes; (void)n_in; (void)out_size; (void)ws_size;
  const float* x       = (const float*)d_in[0];   // 4*512*50*50
  const float* conv1_w = (const float*)d_in[1];   // 512*512*3*3
  const float* conv1_b = (const float*)d_in[2];   // 512
  const float* score_w = (const float*)d_in[3];   // 18*512
  const float* score_b = (const float*)d_in[4];   // 18
  const float* loc_w   = (const float*)d_in[5];   // 36*512
  const float* loc_b   = (const float*)d_in[6];   // 36
  const int*   img_h   = (const int*)d_in[7];
  const int*   img_w   = (const int*)d_in[8];

  float* out = (float*)d_out;
  float* out_locs   = out;            // 360000
  float* out_scores = out + 360000;   // 180000
  float* out_rois   = out + 540000;   // 4800
  int*   out_ridx   = (int*)(out + 544800); // 1200
  float* out_anchor = out + 546000;   // 90000

  // workspace carve-up
  unsigned char* ws = (unsigned char*)d_ws;
  size_t off = 0;
  auto carve = [&](size_t bytes) { size_t r = off; off = (off + bytes + 255) & ~(size_t)255; return r; };
  unsigned short* xbf    = (unsigned short*)(ws + carve((size_t)4 * 2500 * 512 * 2));  // NHWC
  unsigned short* wbf    = (unsigned short*)(ws + carve((size_t)512 * 4608 * 2));      // repacked
  unsigned short* wsl    = (unsigned short*)(ws + carve((size_t)64 * 512 * 2));
  float*          bsl    = (float*)(ws + carve(64 * 4));
  unsigned short* zrow   = (unsigned short*)(ws + carve(256));                         // zero pad
  unsigned short* featbf = (unsigned short*)(ws + carve((size_t)10000 * 512 * 2));     // [m][c]
  float*          head   = (float*)(ws + carve((size_t)10000 * 64 * 4));
  float*          roiws  = (float*)(ws + carve((size_t)4 * 22500 * 4 * 4));
  float*          fgsc   = (float*)(ws + carve((size_t)4 * 22500 * 4));
  float*          skey   = (float*)(ws + carve((size_t)4 * 32768 * 4));
  int*            sidx   = (int*)(ws + carve((size_t)4 * 32768 * 4));

  // 1) layout conversions (NHWC + k-reordered weights), head weight pack, zero row
  k_x_to_nhwc_bf16<<<(4 * 2500 * 512 + 255) / 256, 256, 0, stream>>>(x, xbf);
  k_w_repack_bf16<<<(512 * 4608 + 255) / 256, 256, 0, stream>>>(conv1_w, wbf);
  k_build_head_w<<<(64 * 512 + 255) / 256, 256, 0, stream>>>(score_w, score_b, loc_w, loc_b,
                                                             wsl, bsl, zrow);

  // 2) conv3x3 implicit GEMM (bf16 WMMA, 32x32 wave tiles, async-to-LDS B)
  k_conv3x3_wmma<<<dim3(157, 4), 256, 0, stream>>>(xbf, wbf, conv1_b, zrow, featbf);

  // 3) fused score+loc 1x1 conv GEMM (fully async-to-LDS tiles)
  k_head_wmma<<<157, 256, 0, stream>>>(featbf, wsl, bsl, head);

  // 4) elementwise postprocess (softmax, loc2bbox, clip, filter) + outputs
  k_postproc<<<(4 * 22500 + 255) / 256, 256, 0, stream>>>(head, out_locs, out_scores,
                                                          out_anchor, roiws, fgsc, img_h, img_w);

  // 5) top-6000 per image via bitonic sort (descending), padded to 32768
  k_sort_init<<<(4 * 32768 + 255) / 256, 256, 0, stream>>>(fgsc, skey, sidx);
  for (int k = 2; k <= 32768; k <<= 1)
    for (int j = k >> 1; j > 0; j >>= 1)
      k_bitonic<<<dim3(32768 / 256, 4), 256, 0, stream>>>(skey, sidx, j, k);

  // 6) NMS (boxes in LDS) + compaction to 300 ROIs
  size_t nms_lds = 6000 * sizeof(float4) + 6016;   // ~102KB of the 320KB WGP LDS
  k_nms<<<4, 1024, nms_lds, stream>>>(skey, sidx, roiws, out_rois);

  // 7) roi batch indices
  k_roi_indices<<<5, 256, 0, stream>>>(out_ridx);
}
